// LESSR_part_57604101374706
// MI455X (gfx1250) — matrix operand order repaired and checked by reading, not compiled, not run on hardware
//
#include <hip/hip_runtime.h>
#include <hip/hip_bf16.h>
#include <stdint.h>
#include <stddef.h>

// ----------------------------- model constants -----------------------------
#define N_ITEMS 50000
#define EMB     128
#define BATCH   2048
#define SEQ     32
#define NN      (BATCH*SEQ)     // 65536 nodes
#define DMAX    8
#define BN_EPS  1e-5f

typedef _Float16 h16;
typedef __attribute__((ext_vector_type(16))) _Float16 v16h_t;
typedef __attribute__((ext_vector_type(8)))  _Float16 v8h_t;
typedef __attribute__((ext_vector_type(8)))  float    v8f_t;

// --------------------------- WMMA operand loaders ---------------------------
// A tile (16x32 f16): lane L -> row M = L%16, half = L/16.
// Elements j=0..7  -> K = k0 + half*8 + j ; j=8..15 -> K = k0 + 16 + half*8 + (j-8)
static __device__ __forceinline__ v16h_t lessr_load_a(const h16* __restrict__ abase,
                                                      int k0, int half) {
  const h16* q = abase + k0 + (half << 3);
  union { v16h_t v; v8h_t h[2]; } u;
  u.h[0] = *(const v8h_t*)q;
  u.h[1] = *(const v8h_t*)(q + 16);
  return u.v;
}
// B tile (32x16 f16) read from the LDS stage: lane L -> column n = L%16;
// elements j=0..7 -> K = kstage + half*8 + j ; j=8..15 -> K = kstage + 16 + half*8 + (j-8).
// LDS stage layout: row r (= weight row n0+r) holds 32 contiguous halfs.
static __device__ __forceinline__ v16h_t lessr_load_b_lds(const h16* __restrict__ s,
                                                          int row, int half) {
  const h16* q = s + row * 32 + (half << 3);
  union { v16h_t v; v8h_t h[2]; } u;
  u.h[0] = *(const v8h_t*)q;
  u.h[1] = *(const v8h_t*)(q + 16);
  return u.v;
}

// Stage one 64x32-half tile of W (rows n0..n0+63, cols k0..k0+31) into LDS via
// async copy. 64 rows * 64B = 256 16-byte chunks; 128 threads -> 2 chunks each.
// Out-of-range rows are clamped (their junk only feeds store-guarded C columns).
static __device__ __forceinline__ void lessr_stage_b(const h16* __restrict__ W, int K,
                                                     int Nc, int n0, int k0,
                                                     h16* __restrict__ sbuf, int tid) {
#pragma unroll
  for (int c = tid; c < 256; c += 128) {
    int row = c >> 2;              // 0..63
    int o16 = (c & 3) << 3;        // 0,8,16,24 (halfs)
    int gr = n0 + row; if (gr >= Nc) gr = Nc - 1;
    const h16* g = W + (size_t)gr * (size_t)K + k0 + o16;
    uint32_t lds = (uint32_t)(uintptr_t)(sbuf + row * 32 + o16);
    asm volatile("global_load_async_to_lds_b128 %0, %1, off"
                 :: "v"(lds), "v"((uint64_t)(uintptr_t)g) : "memory");
  }
}

// --------------------------------- GEMM -------------------------------------
// C[M,Nc] (f32, ldc) = gather(A)[M,K] (f16, lda) @ W[Nc,K]^T (f16) + bias (+C).
// Optional row gather: A row for output row r is gidx[r*gstride + goff].
// Block: 128 threads = 4 waves; wave tile 16x64; block tile 64x64.
// B tiles double-buffered in LDS via global_load_async_to_lds_b128 (ASYNCcnt).
// Requires: M % 64 == 0, K % 32 == 0. Nc arbitrary (stores guarded).
__global__ void __launch_bounds__(128)
lessr_gemm_xwT(const h16* __restrict__ A, int lda,
               const int* __restrict__ gidx, int gstride, int goff,
               const h16* __restrict__ W, const float* __restrict__ bias,
               float* __restrict__ C, int ldc,
               int M, int Nc, int K, int accumulate) {
  __shared__ __align__(16) h16 sB[2][64 * 32];   // 2 x 4KB double buffer
  const int tid  = threadIdx.x;
  const int lane = tid & 31;
  const int wave = tid >> 5;
  const int half = lane >> 4;
  const int lm   = lane & 15;
  const int m0 = blockIdx.y * 64 + wave * 16;
  const int n0 = blockIdx.x * 64;

  int ar = m0 + lm;
  if (gidx) ar = gidx[(size_t)ar * (size_t)gstride + goff];
  const h16* abase = A + (size_t)ar * (size_t)lda;

  v8f_t acc[4] = { {}, {}, {}, {} };

  lessr_stage_b(W, K, Nc, n0, 0, &sB[0][0], tid);
  int cur = 0;
  for (int k0 = 0; k0 < K; k0 += 32) {
    if (k0 + 32 < K) {
      lessr_stage_b(W, K, Nc, n0, k0 + 32, &sB[cur ^ 1][0], tid);
      // 2 async instrs just issued (next buf); retire the previous 2 (cur buf).
      asm volatile("s_wait_asynccnt 0x2" ::: "memory");
      __builtin_prefetch(abase + k0 + 64, 0, 0);   // global_prefetch next A line
    } else {
      asm volatile("s_wait_asynccnt 0x0" ::: "memory");
    }
    __syncthreads();                                // all waves' stages visible

    v16h_t a = lessr_load_a(abase, k0, half);
#pragma unroll
    for (int j = 0; j < 4; ++j) {
      v16h_t b = lessr_load_b_lds(&sB[cur][0], j * 16 + lm, half);
      acc[j] = __builtin_amdgcn_wmma_f32_16x16x32_f16(
          /*neg_a=*/false, a, /*neg_b=*/false, b,
          /*c_mod=*/(short)0, acc[j], /*reuse_a=*/false, /*reuse_b=*/false);
    }
    __syncthreads();   // protect sB[cur] from next iteration's stage
    cur ^= 1;
  }

  // Epilogue. C/D layout: lane covers column n = n0 + j*16 + L%16,
  // rows m0 + 8*(L/16) + i for accumulator element i.
#pragma unroll
  for (int j = 0; j < 4; ++j) {
    int n = n0 + j * 16 + lm;
    if (n < Nc) {
      float bj = bias ? bias[n] : 0.0f;
#pragma unroll
      for (int i = 0; i < 8; ++i) {
        size_t off = (size_t)(m0 + half * 8 + i) * (size_t)ldc + (size_t)n;
        float v = acc[j][i] + bj;
        if (accumulate) v += C[off];
        C[off] = v;
      }
    }
  }
}

// ----------------------------- elementwise kernels ---------------------------
__global__ void lessr_renorm(const float* __restrict__ emb,
                             float* __restrict__ embf, h16* __restrict__ embh) {
  int r = blockIdx.x, c = threadIdx.x;            // blockDim = 128 = EMB
  float v = emb[(size_t)r * EMB + c];
  __shared__ float red[EMB];
  red[c] = v * v; __syncthreads();
  for (int o = 64; o > 0; o >>= 1) { if (c < o) red[c] += red[c + o]; __syncthreads(); }
  float nrm = sqrtf(red[0]);
  float s = nrm > 1.0f ? 1.0f / (nrm + 1e-7f) : 1.0f;
  float y = v * s;
  embf[(size_t)r * EMB + c] = y;
  embh[(size_t)r * EMB + c] = (h16)y;
}

__global__ void lessr_cvt(const float* __restrict__ s, h16* __restrict__ d, size_t n) {
  size_t i = (size_t)blockIdx.x * blockDim.x + threadIdx.x;
  if (i < n) d[i] = (h16)s[i];
}

__global__ void lessr_zero(float* __restrict__ a, h16* __restrict__ b, size_t n) {
  size_t i = (size_t)blockIdx.x * blockDim.x + threadIdx.x;
  if (i < n) { a[i] = 0.0f; b[i] = (h16)0.0f; }
}

__global__ void lessr_gather_rows(const float* __restrict__ src, const int* __restrict__ idx,
                                  float* __restrict__ dst, size_t total, int d) {
  size_t i = (size_t)blockIdx.x * blockDim.x + threadIdx.x;
  if (i >= total) return;
  size_t r = i / (size_t)d; int c = (int)(i % (size_t)d);
  dst[i] = src[(size_t)idx[r] * (size_t)d + c];
}

__global__ void lessr_bn_stats(const float* __restrict__ x, int rows, int d,
                               float* __restrict__ mean, float* __restrict__ invstd) {
  int f = blockIdx.x;                              // one feature per block, 256 thr
  float s = 0.0f, q = 0.0f;
  for (int r = threadIdx.x; r < rows; r += 256) {
    float v = x[(size_t)r * (size_t)d + f];
    s += v; q += v * v;
  }
  __shared__ float ss[256], sq[256];
  ss[threadIdx.x] = s; sq[threadIdx.x] = q; __syncthreads();
  for (int o = 128; o > 0; o >>= 1) {
    if ((int)threadIdx.x < o) { ss[threadIdx.x] += ss[threadIdx.x + o];
                                sq[threadIdx.x] += sq[threadIdx.x + o]; }
    __syncthreads();
  }
  if (threadIdx.x == 0) {
    float m = ss[0] / rows;
    float var = sq[0] / rows - m * m;              // biased variance (jnp.var)
    mean[f] = m;
    invstd[f] = rsqrtf(var + BN_EPS);
  }
}

__global__ void lessr_bn_apply(const float* __restrict__ x, const float* __restrict__ mean,
                               const float* __restrict__ invstd, const float* __restrict__ g,
                               const float* __restrict__ b, float* __restrict__ y,
                               h16* __restrict__ yh, size_t total, int d) {
  size_t i = (size_t)blockIdx.x * blockDim.x + threadIdx.x;
  if (i >= total) return;
  int c = (int)(i % (size_t)d);
  float v = (x[i] - mean[c]) * invstd[c] * g[c] + b[c];
  y[i] = v;
  yh[i] = (h16)v;
}

__global__ void lessr_gru_update(const float* __restrict__ gi, const float* __restrict__ gh,
                                 const int* __restrict__ nbr_len, int t,
                                 float* __restrict__ h, h16* __restrict__ hh,
                                 size_t total, int d) {
  size_t i = (size_t)blockIdx.x * blockDim.x + threadIdx.x;
  if (i >= total) return;
  size_t n = i / (size_t)d; int c = (int)(i % (size_t)d);
  size_t b3 = n * (size_t)(3 * d);
  float ir  = gi[b3 + c], iz = gi[b3 + d + c], in_ = gi[b3 + 2 * d + c];
  float hr  = gh[b3 + c], hz = gh[b3 + d + c], hn  = gh[b3 + 2 * d + c];
  float r = 1.0f / (1.0f + __expf(-(ir + hr)));
  float z = 1.0f / (1.0f + __expf(-(iz + hz)));
  float nn = tanhf(in_ + r * hn);
  float hold = h[i];
  float hnew = (1.0f - z) * nn + z * hold;
  float res = (t < nbr_len[n]) ? hnew : hold;
  h[i] = res;
  hh[i] = (h16)res;
}

__global__ void lessr_prelu(float* __restrict__ x, int ld, const float* __restrict__ a,
                            size_t total, int d) {
  size_t i = (size_t)blockIdx.x * blockDim.x + threadIdx.x;
  if (i >= total) return;
  size_t r = i / (size_t)d; int c = (int)(i % (size_t)d);
  size_t off = r * (size_t)ld + c;
  float v = x[off];
  x[off] = v > 0.0f ? v : a[c] * v;
}

__global__ void lessr_copy_cols(const float* __restrict__ src, int lds_,
                                float* __restrict__ dst, int ldd,
                                size_t total, int cols) {
  size_t i = (size_t)blockIdx.x * blockDim.x + threadIdx.x;
  if (i >= total) return;
  size_t r = i / (size_t)cols; int c = (int)(i % (size_t)cols);
  dst[r * (size_t)ldd + c] = src[r * (size_t)lds_ + c];
}

__global__ void lessr_gather_last_h(const h16* __restrict__ src, h16* __restrict__ dst,
                                    int S, int d, size_t total) {
  size_t i = (size_t)blockIdx.x * blockDim.x + threadIdx.x;
  if (i >= total) return;
  size_t b = i / (size_t)d; int c = (int)(i % (size_t)d);
  dst[i] = src[((size_t)b * S + (S - 1)) * (size_t)d + c];
}

__global__ void lessr_gather_last_f32(const float* __restrict__ src, float* __restrict__ dst,
                                      int ldd, int S, int d, size_t total) {
  size_t i = (size_t)blockIdx.x * blockDim.x + threadIdx.x;
  if (i >= total) return;
  size_t b = i / (size_t)d; int c = (int)(i % (size_t)d);
  dst[b * (size_t)ldd + c] = src[((size_t)b * S + (S - 1)) * (size_t)d + c];
}

// e[n] = sum_k sigmoid(fu[n,k] + fv[n/S,k]) * we[k]   (one row per block, 128 thr)
__global__ void lessr_attn(const float* __restrict__ fu, const float* __restrict__ fv,
                           const float* __restrict__ we, float* __restrict__ e, int S) {
  int n = blockIdx.x, k = threadIdx.x;
  int b = n / S;
  float v = fu[(size_t)n * EMB + k] + fv[(size_t)b * EMB + k];
  v = (1.0f / (1.0f + __expf(-v))) * we[k];
  __shared__ float red[EMB];
  red[k] = v; __syncthreads();
  for (int o = 64; o > 0; o >>= 1) { if (k < o) red[k] += red[k + o]; __syncthreads(); }
  if (k == 0) e[n] = red[0];
}

// softmax over S=32 per session (block = 32 threads = 1 wave)
__global__ void lessr_softmax(const float* __restrict__ e, float* __restrict__ beta) {
  int b = blockIdx.x, s = threadIdx.x;
  float v = e[(size_t)b * SEQ + s];
  __shared__ float sh[SEQ];
  sh[s] = v; __syncthreads();
  for (int o = 16; o > 0; o >>= 1) { if (s < o) sh[s] = fmaxf(sh[s], sh[s + o]); __syncthreads(); }
  float m = sh[0]; __syncthreads();
  float ex = __expf(v - m);
  sh[s] = ex; __syncthreads();
  for (int o = 16; o > 0; o >>= 1) { if (s < o) sh[s] += sh[s + o]; __syncthreads(); }
  beta[(size_t)b * SEQ + s] = ex / sh[0];
}

// out[b,c] = sum_s beta[b,s] * fr[b*S+s, c]
__global__ void lessr_wsum(const float* __restrict__ beta, const float* __restrict__ fr,
                           float* __restrict__ out, int S, int d, size_t total) {
  size_t i = (size_t)blockIdx.x * blockDim.x + threadIdx.x;
  if (i >= total) return;
  size_t b = i / (size_t)d; int c = (int)(i % (size_t)d);
  float acc = 0.0f;
  for (int s = 0; s < S; ++s)
    acc += beta[b * S + s] * fr[((size_t)b * S + s) * (size_t)d + c];
  out[i] = acc;
}

// ------------------------------- host driver --------------------------------
static inline unsigned cdiv(size_t a, size_t b) { return (unsigned)((a + b - 1) / b); }

extern "C" void kernel_launch(void* const* d_in, const int* in_sizes, int n_in,
                              void* d_out, int out_size, void* d_ws, size_t ws_size,
                              hipStream_t stream) {
  (void)in_sizes; (void)n_in; (void)out_size; (void)ws_size;

  // ---- input map (JAX pytree order: dict keys sorted alphabetically) ----
  const int*   iid      = (const int*)  d_in[0];   // [N]
  const int*   nbr_idx  = (const int*)  d_in[1];   // [N, DMAX]
  const int*   nbr_len  = (const int*)  d_in[2];   // [N]
  const float* top_bn_b = (const float*)d_in[3];   // [512]
  const float* top_bn_g = (const float*)d_in[4];   // [512]
  const float* emb      = (const float*)d_in[5];   // [50000,128]
  const float *L_bhh[2], *L_bih[2], *L_bn_b[2], *L_bn_g[2], *L_prelu[2],
              *L_wneigh[2], *L_wself[2], *L_whh[2], *L_wih[2];
  for (int l = 0; l < 2; ++l) {
    int b0 = 6 + 9 * l;  // per-layer keys sorted: bhh,bih,bn_b,bn_g,prelu_a,w_neigh,w_self,whh,wih
    L_bhh[l]    = (const float*)d_in[b0 + 0];
    L_bih[l]    = (const float*)d_in[b0 + 1];
    L_bn_b[l]   = (const float*)d_in[b0 + 2];
    L_bn_g[l]   = (const float*)d_in[b0 + 3];
    L_prelu[l]  = (const float*)d_in[b0 + 4];
    L_wneigh[l] = (const float*)d_in[b0 + 5];
    L_wself[l]  = (const float*)d_in[b0 + 6];
    L_whh[l]    = (const float*)d_in[b0 + 7];
    L_wih[l]    = (const float*)d_in[b0 + 8];
  }
  // ro keys sorted: bn_b,bn_g,bv,prelu_a,we,wout,wu,wv
  const float* ro_bn_b  = (const float*)d_in[24];
  const float* ro_bn_g  = (const float*)d_in[25];
  const float* ro_bv    = (const float*)d_in[26];
  const float* ro_prelu = (const float*)d_in[27];
  const float* ro_we    = (const float*)d_in[28];  // [1,128]
  const float* ro_wout  = (const float*)d_in[29];  // [128,384]
  const float* ro_wu    = (const float*)d_in[30];  // [128,384]
  const float* ro_wv    = (const float*)d_in[31];  // [128,384]
  const float* w_sr     = (const float*)d_in[32];  // [128,512]

  // ---- workspace suballocation ----
  char* wsp = (char*)d_ws; size_t off = 0;
  auto alloc = [&](size_t bytes) -> void* {
    size_t a = (off + 255) & ~(size_t)255; off = a + bytes; return (void*)(wsp + a);
  };
  float* emb_f   = (float*)alloc(sizeof(float) * (size_t)N_ITEMS * EMB);
  h16*   emb_h   = (h16*)  alloc(sizeof(h16)   * (size_t)N_ITEMS * EMB);
  float* feat0   = (float*)alloc(sizeof(float) * (size_t)NN * 128);
  float* feat1   = (float*)alloc(sizeof(float) * (size_t)NN * 256);
  float* feat2   = (float*)alloc(sizeof(float) * (size_t)NN * 384);
  float* fnorm   = (float*)alloc(sizeof(float) * (size_t)NN * 384);
  h16*   fnorm_h = (h16*)  alloc(sizeof(h16)   * (size_t)NN * 384);
  float* hbuf    = (float*)alloc(sizeof(float) * (size_t)NN * 256);
  h16*   hh      = (h16*)  alloc(sizeof(h16)   * (size_t)NN * 256);
  float* gi      = (float*)alloc(sizeof(float) * (size_t)NN * 768);
  float* gh      = (float*)alloc(sizeof(float) * (size_t)NN * 768);
  float* fu      = (float*)alloc(sizeof(float) * (size_t)NN * 128);
  h16*   frl_h   = (h16*)  alloc(sizeof(h16)   * (size_t)BATCH * 384);
  float* fv      = (float*)alloc(sizeof(float) * (size_t)BATCH * 128);
  float* ebuf    = (float*)alloc(sizeof(float) * (size_t)NN);
  float* beta    = (float*)alloc(sizeof(float) * (size_t)NN);
  float* srg     = (float*)alloc(sizeof(float) * (size_t)BATCH * 384);
  h16*   srg_h   = (h16*)  alloc(sizeof(h16)   * (size_t)BATCH * 384);
  float* sr_cat  = (float*)alloc(sizeof(float) * (size_t)BATCH * 512);
  float* sr_bn   = (float*)alloc(sizeof(float) * (size_t)BATCH * 512);
  h16*   sr_bn_h = (h16*)  alloc(sizeof(h16)   * (size_t)BATCH * 512);
  h16*   sr_h    = (h16*)  alloc(sizeof(h16)   * (size_t)BATCH * 128);
  float* mean    = (float*)alloc(sizeof(float) * 512);
  float* invstd  = (float*)alloc(sizeof(float) * 512);
  // f16 weight copies
  h16* wih_h[2]    = { (h16*)alloc(2 * (size_t)384 * 128), (h16*)alloc(2 * (size_t)768 * 256) };
  h16* whh_h[2]    = { (h16*)alloc(2 * (size_t)384 * 128), (h16*)alloc(2 * (size_t)768 * 256) };
  h16* wself_h[2]  = { (h16*)alloc(2 * (size_t)128 * 128), (h16*)alloc(2 * (size_t)128 * 256) };
  h16* wneigh_h[2] = { (h16*)alloc(2 * (size_t)128 * 128), (h16*)alloc(2 * (size_t)128 * 256) };
  h16* wu_h   = (h16*)alloc(2 * (size_t)128 * 384);
  h16* wv_h   = (h16*)alloc(2 * (size_t)128 * 384);
  h16* wout_h = (h16*)alloc(2 * (size_t)128 * 384);
  h16* wsr_h  = (h16*)alloc(2 * (size_t)128 * 512);

  const unsigned TB = 256;
  auto cvt = [&](const float* s, h16* d, size_t n) {
    lessr_cvt<<<cdiv(n, TB), TB, 0, stream>>>(s, d, n);
  };
  // gemm with optional A-row gather (gidx != nullptr -> row = gidx[r*gstride+goff])
  auto gemm = [&](const h16* A, int lda, const int* gidx, int gstride, int goff,
                  const h16* W, const float* bias,
                  float* C, int ldc, int M, int Nc, int K, int acc) {
    dim3 g((unsigned)((Nc + 63) / 64), (unsigned)(M / 64));
    lessr_gemm_xwT<<<g, 128, 0, stream>>>(A, lda, gidx, gstride, goff,
                                          W, bias, C, ldc, M, Nc, K, acc);
  };

  // ---- embedding renorm + f16 copies of all GEMM weights ----
  lessr_renorm<<<N_ITEMS, EMB, 0, stream>>>(emb, emb_f, emb_h);
  cvt(L_wih[0],    wih_h[0],    (size_t)384 * 128);
  cvt(L_whh[0],    whh_h[0],    (size_t)384 * 128);
  cvt(L_wself[0],  wself_h[0],  (size_t)128 * 128);
  cvt(L_wneigh[0], wneigh_h[0], (size_t)128 * 128);
  cvt(L_wih[1],    wih_h[1],    (size_t)768 * 256);
  cvt(L_whh[1],    whh_h[1],    (size_t)768 * 256);
  cvt(L_wself[1],  wself_h[1],  (size_t)128 * 256);
  cvt(L_wneigh[1], wneigh_h[1], (size_t)128 * 256);
  cvt(ro_wu,   wu_h,   (size_t)128 * 384);
  cvt(ro_wv,   wv_h,   (size_t)128 * 384);
  cvt(ro_wout, wout_h, (size_t)128 * 384);
  cvt(w_sr,    wsr_h,  (size_t)128 * 512);

  // ---- feat0 = renormed_emb[iid] ----
  lessr_gather_rows<<<cdiv((size_t)NN * 128, TB), TB, 0, stream>>>(
      emb_f, iid, feat0, (size_t)NN * 128, 128);

  // ---- EOPA layers ----
  float* feats[3] = { feat0, feat1, feat2 };
  for (int l = 0; l < 2; ++l) {
    const int d = EMB * (l + 1), d3 = 3 * d, dn = d + EMB;
    float* fin  = feats[l];
    float* fout = feats[l + 1];
    const size_t nd = (size_t)NN * d;

    lessr_bn_stats<<<d, 256, 0, stream>>>(fin, NN, d, mean, invstd);
    lessr_bn_apply<<<cdiv(nd, TB), TB, 0, stream>>>(
        fin, mean, invstd, L_bn_g[l], L_bn_b[l], fnorm, fnorm_h, nd, d);

    lessr_zero<<<cdiv(nd, TB), TB, 0, stream>>>(hbuf, hh, nd);
    for (int t = 0; t < DMAX; ++t) {
      // gi = f[nbr_idx[:,t]] @ wih^T + bih   (gather fused into GEMM A-load)
      gemm(fnorm_h, d, nbr_idx, DMAX, t, wih_h[l], L_bih[l], gi, d3, NN, d3, d, 0);
      // gh = h @ whh^T + bhh
      gemm(hh, d, nullptr, 0, 0, whh_h[l], L_bhh[l], gh, d3, NN, d3, d, 0);
      lessr_gru_update<<<cdiv(nd, TB), TB, 0, stream>>>(gi, gh, nbr_len, t, hbuf, hh, nd, d);
    }
    // out = prelu(f @ w_self^T + neigh @ w_neigh^T); write into cols [0,128) of fout
    gemm(fnorm_h, d, nullptr, 0, 0, wself_h[l],  nullptr, fout, dn, NN, EMB, d, 0);
    gemm(hh,      d, nullptr, 0, 0, wneigh_h[l], nullptr, fout, dn, NN, EMB, d, 1);
    lessr_prelu<<<cdiv((size_t)NN * EMB, TB), TB, 0, stream>>>(
        fout, dn, L_prelu[l], (size_t)NN * EMB, EMB);
    // feat_next = [out | feat] : copy old feat into cols [128, 128+d)
    lessr_copy_cols<<<cdiv(nd, TB), TB, 0, stream>>>(fin, d, fout + EMB, dn, nd, d);
  }

  // ---- readout ----
  const int din = 384;
  lessr_bn_stats<<<din, 256, 0, stream>>>(feat2, NN, din, mean, invstd);
  lessr_bn_apply<<<cdiv((size_t)NN * din, TB), TB, 0, stream>>>(
      feat2, mean, invstd, ro_bn_g, ro_bn_b, fnorm, fnorm_h, (size_t)NN * din, din);

  gemm(fnorm_h, din, nullptr, 0, 0, wu_h, nullptr, fu, EMB, NN, EMB, din, 0);   // fu
  lessr_gather_last_h<<<cdiv((size_t)BATCH * din, TB), TB, 0, stream>>>(
      fnorm_h, frl_h, SEQ, din, (size_t)BATCH * din);
  gemm(frl_h, din, nullptr, 0, 0, wv_h, ro_bv, fv, EMB, BATCH, EMB, din, 0);    // fv

  lessr_attn<<<NN, EMB, 0, stream>>>(fu, fv, ro_we, ebuf, SEQ);
  lessr_softmax<<<BATCH, SEQ, 0, stream>>>(ebuf, beta);
  lessr_wsum<<<cdiv((size_t)BATCH * din, TB), TB, 0, stream>>>(
      beta, fnorm, srg, SEQ, din, (size_t)BATCH * din);

  cvt(srg, srg_h, (size_t)BATCH * din);
  // sr_g = prelu(srg @ wout^T) -> cols [384,512) of sr_cat ; sr_l -> cols [0,384)
  gemm(srg_h, din, nullptr, 0, 0, wout_h, nullptr, sr_cat + din, 512, BATCH, EMB, din, 0);
  lessr_prelu<<<cdiv((size_t)BATCH * EMB, TB), TB, 0, stream>>>(
      sr_cat + din, 512, ro_prelu, (size_t)BATCH * EMB, EMB);
  lessr_gather_last_f32<<<cdiv((size_t)BATCH * din, TB), TB, 0, stream>>>(
      feat2, sr_cat, 512, SEQ, din, (size_t)BATCH * din);

  lessr_bn_stats<<<512, 256, 0, stream>>>(sr_cat, BATCH, 512, mean, invstd);
  lessr_bn_apply<<<cdiv((size_t)BATCH * 512, TB), TB, 0, stream>>>(
      sr_cat, mean, invstd, top_bn_g, top_bn_b, sr_bn, sr_bn_h, (size_t)BATCH * 512, 512);

  float* out_sr     = (float*)d_out;                    // [2048,128]
  float* out_logits = out_sr + (size_t)BATCH * EMB;     // [2048,50000]
  gemm(sr_bn_h, 512, nullptr, 0, 0, wsr_h, nullptr, out_sr, EMB, BATCH, EMB, 512, 0);
  cvt(out_sr, sr_h, (size_t)BATCH * EMB);
  gemm(sr_h, EMB, nullptr, 0, 0, emb_h, nullptr, out_logits, N_ITEMS, BATCH, N_ITEMS, EMB, 0);
}
